// LSTM_51196010168691
// MI455X (gfx1250) — matrix operand order repaired and checked by reading, not compiled, not run on hardware
//
#include <hip/hip_runtime.h>

// Problem dims (match reference)
constexpr int Bc = 64;
constexpr int Sc = 512;
constexpr int Hc = 1024;
constexpr int Mc = Bc * Sc;     // 32768 GEMM rows
constexpr int Kc = Hc;          // 1024
constexpr int Nc = 4 * Hc;      // 4096

typedef __attribute__((ext_vector_type(16))) __bf16 v16bf;
typedef __attribute__((ext_vector_type(8)))  __bf16 v8bf;
typedef __attribute__((ext_vector_type(8)))  float  v8f;

union BF16Frag { v16bf v; v8bf h[2]; };

struct Frags {
    BF16Frag a[2];   // two 16x32 A tiles (rows m0..m0+15, m0+16..m0+31)
    BF16Frag b[4];   // four 32x16 B tiles (cols n0 .. n0+63)
};

static __device__ __forceinline__ unsigned short f32_to_bf16_bits(float f) {
    unsigned int u = __float_as_uint(f);
    unsigned int rnd = 0x7FFFu + ((u >> 16) & 1u);   // round-to-nearest-even
    return (unsigned short)((u + rnd) >> 16);
}

static __device__ __forceinline__ float sigmoidf_fast(float x) {
    return 1.0f / (1.0f + __expf(-x));
}

// ---------------------------------------------------------------------------
// Kernel 1: fp32 x -> bf16 (vectorized: float4 in, 8-byte packed out)
// ---------------------------------------------------------------------------
__global__ void convert_x_bf16(const float* __restrict__ x,
                               unsigned short* __restrict__ xb) {
    size_t i = ((size_t)blockIdx.x * blockDim.x + threadIdx.x) * 4;
    float4 v = *reinterpret_cast<const float4*>(x + i);
    union { unsigned short u[4]; uint2 d; } o;
    o.u[0] = f32_to_bf16_bits(v.x);
    o.u[1] = f32_to_bf16_bits(v.y);
    o.u[2] = f32_to_bf16_bits(v.z);
    o.u[3] = f32_to_bf16_bits(v.w);
    *reinterpret_cast<uint2*>(xb + i) = o.d;
}

// ---------------------------------------------------------------------------
// Kernel 2: U [K,N] fp32 -> Ut [N,K] bf16, LDS 32x33 tile transpose
// ---------------------------------------------------------------------------
__global__ void transpose_convert_U(const float* __restrict__ U,
                                    unsigned short* __restrict__ Ut) {
    __shared__ float tile[32][33];
    const int nBlock = blockIdx.x * 32;     // along N
    const int kBlock = blockIdx.y * 32;     // along K
    const int tx = threadIdx.x & 31;
    const int ty = threadIdx.x >> 5;        // 0..7
#pragma unroll
    for (int i = ty; i < 32; i += 8)
        tile[i][tx] = U[(size_t)(kBlock + i) * Nc + (nBlock + tx)];
    __syncthreads();
#pragma unroll
    for (int i = ty; i < 32; i += 8)
        Ut[(size_t)(nBlock + i) * Kc + (kBlock + tx)] = f32_to_bf16_bits(tile[tx][i]);
}

// ---------------------------------------------------------------------------
// Kernel 3: GEMM gates[M,N] = Xb[M,K] @ Ut[N,K]^T via v_wmma_f32_16x16x32_bf16
// Block: 256 threads = 8 waves arranged 2(M) x 4(N) -> 64 x 256 block tile.
// Wave tile: 32(M) x 64(N) = 8 WMMA accumulators.
// Ping-pong fragment buffers with #pragma unroll 1 on the k-loop: loads of one
// buffer overlap WMMAs on the other, while live VGPRs stay < 256 (no spills).
// ---------------------------------------------------------------------------
static __device__ __forceinline__ void load_frags(const unsigned short* __restrict__ pA0,
                                                  const unsigned short* __restrict__ pA1,
                                                  const unsigned short* __restrict__ pB0,
                                                  int k, Frags& f) {
    f.a[0].h[0] = *reinterpret_cast<const v8bf*>(pA0 + k);
    f.a[0].h[1] = *reinterpret_cast<const v8bf*>(pA0 + k + 16);
    f.a[1].h[0] = *reinterpret_cast<const v8bf*>(pA1 + k);
    f.a[1].h[1] = *reinterpret_cast<const v8bf*>(pA1 + k + 16);
#pragma unroll
    for (int nt = 0; nt < 4; ++nt) {
        const unsigned short* pB = pB0 + (size_t)(nt * 16) * Kc + k;
        f.b[nt].h[0] = *reinterpret_cast<const v8bf*>(pB);
        f.b[nt].h[1] = *reinterpret_cast<const v8bf*>(pB + 16);
    }
}

static __device__ __forceinline__ void mma_step(const Frags& f, v8f acc[2][4]) {
#pragma unroll
    for (int mi = 0; mi < 2; ++mi)
#pragma unroll
        for (int nt = 0; nt < 4; ++nt)
            acc[mi][nt] = __builtin_amdgcn_wmma_f32_16x16x32_bf16(
                false, f.a[mi].v, false, f.b[nt].v, (short)0, acc[mi][nt],
                false, false);
}

__global__ void __launch_bounds__(256)
gemm_wmma_bf16(const unsigned short* __restrict__ Xb,
               const unsigned short* __restrict__ Ut,
               float* __restrict__ gates) {
    const int lane   = threadIdx.x & 31;
    const int wave   = threadIdx.x >> 5;     // 0..7
    const int laneLo = lane & 15;
    const int laneHi = lane >> 4;            // 0 or 1
    const int waveM  = wave & 1;             // 2 M-tiles per block (32 rows each)
    const int waveN  = wave >> 1;            // 4 N-tiles per block (64 cols each)
    const int m0 = blockIdx.y * 64 + waveM * 32;
    const int n0 = blockIdx.x * 256 + waveN * 64;

    v8f acc[2][4] = {{v8f{}, v8f{}, v8f{}, v8f{}},
                     {v8f{}, v8f{}, v8f{}, v8f{}}};

    // A fragment addressing (16-bit A 16x32 layout):
    //   lanes 0-15 : row = m, K = k..k+7 and k+16..k+23
    //   lanes 16-31: row = m, K = k+8..k+15 and k+24..k+31
    const unsigned short* pA0 = Xb + (size_t)(m0 + laneLo) * Kc + laneHi * 8;
    const unsigned short* pA1 = pA0 + (size_t)16 * Kc;

    // B fragment addressing (16-bit B 32x16 layout):
    //   lanes 0-15 : col = n, K = k..k+7 and k+16..k+23
    //   lanes 16-31: col = n, K = k+8..k+15 and k+24..k+31
    const unsigned short* pB0 = Ut + (size_t)(n0 + laneLo) * Kc + laneHi * 8;

    // Software pipeline: explicit ping-pong, unroll factor locked at 2.
    Frags f0, f1;
    load_frags(pA0, pA1, pB0, 0, f0);
#pragma unroll 1
    for (int k = 0; k < Kc - 64; k += 64) {
        load_frags(pA0, pA1, pB0, k + 32, f1);   // overlaps mma on f0
        mma_step(f0, acc);
        load_frags(pA0, pA1, pB0, k + 64, f0);   // overlaps mma on f1
        mma_step(f1, acc);
    }
    load_frags(pA0, pA1, pB0, Kc - 32, f1);      // last prefetch
    mma_step(f0, acc);
    mma_step(f1, acc);

    // C/D f32 16x16 layout: VGPR r -> row base + r + 8*laneHi, col = tile + laneLo
#pragma unroll
    for (int mi = 0; mi < 2; ++mi) {
#pragma unroll
        for (int nt = 0; nt < 4; ++nt) {
            const int col = n0 + nt * 16 + laneLo;
            float* pC = gates + (size_t)(m0 + mi * 16 + laneHi * 8) * Nc + col;
#pragma unroll
            for (int r = 0; r < 8; ++r)
                pC[(size_t)r * Nc] = acc[mi][nt][r];
        }
    }
}

// ---------------------------------------------------------------------------
// Kernel 4: elementwise LSTM recurrence. One thread per (b, h) channel.
// gates don't depend on h_{t-1}, so all B*H channels are independent.
// out layout: hidden[B,S,H] | h_T[B,H] | c_T[B,H]
// ---------------------------------------------------------------------------
__global__ void lstm_recurrence(const float* __restrict__ gates,
                                const float* __restrict__ bias,
                                float* __restrict__ out) {
    const int tid = blockIdx.x * blockDim.x + threadIdx.x;  // 0 .. B*H-1
    const int b = tid >> 10;        // / Hc
    const int h = tid & (Hc - 1);   // % Hc

    const float bi = bias[h];
    const float bf = bias[Hc + h];
    const float bg = bias[2 * Hc + h];
    const float bo = bias[3 * Hc + h];

    const float* g = gates + (size_t)b * Sc * Nc + h;
    float* hseq = out + (size_t)b * Sc * Hc + h;

    float c = 0.0f, hv = 0.0f;
    for (int t = 0; t < Sc; ++t) {
        const float* gr = g + (size_t)t * Nc;
        float it = sigmoidf_fast(gr[0]      + bi);
        float ft = sigmoidf_fast(gr[Hc]     + bf);
        float gt = tanhf(        gr[2 * Hc] + bg);
        float ot = sigmoidf_fast(gr[3 * Hc] + bo);
        c  = ft * c + it * gt;
        hv = ot * tanhf(c);
        hseq[(size_t)t * Hc] = hv;
    }
    const size_t hiddenElems = (size_t)Bc * Sc * Hc;
    out[hiddenElems + (size_t)b * Hc + h]                   = hv;  // h_T
    out[hiddenElems + (size_t)Bc * Hc + (size_t)b * Hc + h] = c;   // c_T
}

// ---------------------------------------------------------------------------
extern "C" void kernel_launch(void* const* d_in, const int* in_sizes, int n_in,
                              void* d_out, int out_size, void* d_ws, size_t ws_size,
                              hipStream_t stream) {
    (void)in_sizes; (void)n_in; (void)out_size; (void)ws_size;

    const float* x    = (const float*)d_in[0];   // [B,S,H] fp32
    const float* U    = (const float*)d_in[1];   // [H,4H]  fp32
    const float* bias = (const float*)d_in[2];   // [4H]    fp32
    float* out = (float*)d_out;

    // Workspace: Xb bf16 [M*K] (64MB) | Ut bf16 [N*K] (8MB) | gates f32 [M*N] (512MB)
    unsigned short* Xb = (unsigned short*)d_ws;
    unsigned short* Ut = Xb + (size_t)Mc * Kc;
    float* gates = (float*)(Ut + (size_t)Nc * Kc);

    // 1) x -> bf16
    {
        size_t nThreads = ((size_t)Mc * Kc) / 4;          // float4 per thread
        convert_x_bf16<<<(unsigned)(nThreads / 256), 256, 0, stream>>>(x, Xb);
    }
    // 2) U -> bf16, transposed to [N,K]
    {
        dim3 grid(Nc / 32, Kc / 32);
        transpose_convert_U<<<grid, 256, 0, stream>>>(U, Ut);
    }
    // 3) GEMM: gates = X @ U   (bias folded in during recurrence)
    {
        dim3 grid(Nc / 256, Mc / 64);
        gemm_wmma_bf16<<<grid, 256, 0, stream>>>(Xb, Ut, gates);
    }
    // 4) Recurrence + outputs
    {
        lstm_recurrence<<<(Bc * Hc) / 256, 256, 0, stream>>>(gates, bias, out);
    }
}
